// DFlashAttention_83846351552861
// MI455X (gfx1250) — matrix operand order repaired and checked by reading, not compiled, not run on hardware
//
#include <hip/hip_runtime.h>
#include <hip/hip_bf16.h>

#define DEV __device__ __forceinline__

typedef __attribute__((ext_vector_type(16))) __bf16 bf16x16;
typedef __attribute__((ext_vector_type(8)))  float  f32x8;

// ---- problem constants (from reference) ----
constexpr int B_ = 4, S_ = 128, HID_ = 4096;
constexpr int D_ = 128, HQ_ = 32, HK_ = 8;
constexpr int PSIZE_ = 256, NPP_ = 16;
constexpr int LCACHE = NPP_ * PSIZE_;   // 4096 cached keys per batch
constexpr float SCALE_ = 0.08838834764831845f; // 1/sqrt(128)

// LDS tile strides (bf16 elements). Both are 16B multiples; bank-skewed.
constexpr int KSTR = 136;  // 272B per K row   (32 rows)
constexpr int VSTR = 40;   // 80B  per V^T row (128 rows)

DEV f32x8 zero_f32x8() {
  f32x8 v;
#pragma unroll
  for (int i = 0; i < 8; ++i) v[i] = 0.f;
  return v;
}

// ---- WMMA fragment loaders (wave32, 16-bit operands in contiguous k = 16*(lane/16) + s order for both A and B) ----
// A (16x32 bf16), row-major K-contiguous source. lanes 0-15: K 0-15;
// lanes 16-31: K 16-31; row = lane%16.
DEV bf16x16 load_frag_a_bf(const __bf16* src, int ldk) {
  const int lane = threadIdx.x & 31;
  const int ln = lane & 15, hf = lane >> 4;
  const unsigned int* p = (const unsigned int*)(src + (size_t)ln * ldk);
  union { bf16x16 v; unsigned int d[8]; } u;
  const int b0 = hf * 8;
#pragma unroll
  for (int t = 0; t < 4; ++t) {
    u.d[t]     = p[b0 + t];       // K = hf*16 + 0..7
    u.d[4 + t] = p[b0 + 4 + t];   // K = hf*16 + 8..15
  }
  return u.v;
}

// Same A layout, element-wise (for LDS P tile: load type == store type).
DEV bf16x16 load_frag_a_lds(const __bf16* src, int ldk) {
  const int lane = threadIdx.x & 31;
  const int ln = lane & 15, hf = lane >> 4;
  const __bf16* p = src + (size_t)ln * ldk;
  union { bf16x16 v; __bf16 e[16]; } u;
#pragma unroll
  for (int j = 0; j < 8; ++j) {
    u.e[j]     = p[hf * 16 + j];
    u.e[8 + j] = p[hf * 16 + 8 + j];
  }
  return u.v;
}

// B (32x16 bf16): lane%16 = N column; lane half selects K in [0,16)/[16,32).
// Source stored as N rows x K contiguous.
DEV bf16x16 load_frag_b_bf(const __bf16* src, int ldk) {
  const int lane = threadIdx.x & 31;
  const int ln = lane & 15, hf = lane >> 4;
  const unsigned int* p = (const unsigned int*)(src + (size_t)ln * ldk);
  union { bf16x16 v; unsigned int d[8]; } u;
#pragma unroll
  for (int t = 0; t < 8; ++t) u.d[t] = p[hf * 8 + t];
  return u.v;
}

// 16 contiguous bf16 from a 16B-aligned pointer (two b128 loads).
DEV bf16x16 load16_aligned(const __bf16* p) {
  union { bf16x16 v; uint4 q[2]; } u;
  const uint4* s = (const uint4*)p;
  u.q[0] = s[0];
  u.q[1] = s[1];
  return u.v;
}

// ---- elementwise fp32 -> bf16 ----
__global__ void cvt_f32_to_bf16(const float* __restrict__ in, __bf16* __restrict__ out, int n) {
  int i = blockIdx.x * blockDim.x + threadIdx.x;
  if (i < n) out[i] = (__bf16)in[i];
}

// ---- C[M,N] = A[M,K] * B[N,K]^T, bf16 in, fp32 out ----
// grid (N/256, M/16), block 128 (4 waves); each wave owns a 16x64 strip.
// All 4 B fragments are loaded before the 4 WMMAs so the loads clause together
// and the matrix pipe runs back-to-back instead of stalling per tile.
__global__ void __launch_bounds__(128)
gemm_bf16_wmma(const __bf16* __restrict__ A, const __bf16* __restrict__ Bm,
               float* __restrict__ C, int M, int N, int K) {
  const int wave = threadIdx.x >> 5;
  const int m0 = blockIdx.y * 16;
  const int n0 = blockIdx.x * 256 + wave * 64;
  f32x8 acc[4];
#pragma unroll
  for (int i = 0; i < 4; ++i) acc[i] = zero_f32x8();
  const __bf16* arow = A + (size_t)m0 * K;
  for (int k0 = 0; k0 < K; k0 += 32) {
    const bf16x16 af = load_frag_a_bf(arow + k0, K);
    bf16x16 bfr[4];
#pragma unroll
    for (int i = 0; i < 4; ++i)
      bfr[i] = load_frag_b_bf(Bm + (size_t)(n0 + 16 * i) * K + k0, K);
#pragma unroll
    for (int i = 0; i < 4; ++i)
      acc[i] = __builtin_amdgcn_wmma_f32_16x16x32_bf16(false, af, false, bfr[i], (short)0,
                                                       acc[i], false, false);
  }
  const int lane = threadIdx.x & 31, ln = lane & 15, hf = lane >> 4;
#pragma unroll
  for (int i = 0; i < 4; ++i)
#pragma unroll
    for (int r = 0; r < 8; ++r)
      C[(size_t)(m0 + r + 8 * hf) * N + n0 + 16 * i + ln] = acc[i][r];
}

// ---- RMSNorm + RoPE over one (b,s,h) row of D=128; one wave per row ----
__global__ void __launch_bounds__(32)
rmsnorm_rope_kernel(const float* __restrict__ tin, const float* __restrict__ w,
                    const int* __restrict__ seqlens, __bf16* __restrict__ tout, int H) {
  const int h = blockIdx.x, s = blockIdx.y, b = blockIdx.z;
  const int t = threadIdx.x;  // 0..31 (wave32)
  const size_t base = (((size_t)b * S_ + s) * H + h) * (size_t)D_;
  const float* row = tin + base;
  float a0 = row[t], a1 = row[t + 32], b0 = row[t + 64], b1 = row[t + 96];
  float ssq = a0 * a0 + a1 * a1 + b0 * b0 + b1 * b1;
#pragma unroll
  for (int m = 16; m >= 1; m >>= 1) ssq += __shfl_xor(ssq, m, 32);
  const float rinv = rsqrtf(ssq * (1.f / 128.f) + 1e-6f);
  a0 *= rinv * w[t];      a1 *= rinv * w[t + 32];
  b0 *= rinv * w[t + 64]; b1 *= rinv * w[t + 96];
  const float pos = (float)(seqlens[b] + s);
  const float f0 = pos * __powf(10000.f, -((float)t) * (1.f / 64.f));
  const float f1 = pos * __powf(10000.f, -((float)(t + 32)) * (1.f / 64.f));
  float sn0, cs0, sn1, cs1;
  __sincosf(f0, &sn0, &cs0);
  __sincosf(f1, &sn1, &cs1);
  __bf16* out = tout + base;
  out[t]      = (__bf16)(a0 * cs0 - b0 * sn0);
  out[t + 64] = (__bf16)(b0 * cs0 + a0 * sn0);
  out[t + 32] = (__bf16)(a1 * cs1 - b1 * sn1);
  out[t + 96] = (__bf16)(b1 * cs1 + a1 * sn1);
}

// ---- flash attention over paged KV cache (bf16) + new K/V, GQA rep=4 ----
// grid (HQ, B), block 256 = 8 waves; wave w handles query rows [16w, 16w+16).
// Valid key blocks are remapped to a dense index; K (async-LDS DMA) and V^T
// (VGPR transpose) are double-buffered so block i+1 streams in while block i
// is computed. One barrier per block.
__global__ void __launch_bounds__(256)
attn_kernel(const __bf16* __restrict__ qbf,   // (B,S,HQ,D)
            const __bf16* __restrict__ knew,  // (B,S,HK,D)
            const __bf16* __restrict__ vnew,  // (B,S,HK,D)
            const __bf16* __restrict__ kcbf,  // (PAGES,PSIZE,HK,D) bf16
            const __bf16* __restrict__ vcbf,
            const int* __restrict__ btab,     // (B,NPP)
            const int* __restrict__ seqlens,  // (B)
            __bf16* __restrict__ obf) {       // (B,S,HQ,D)
  __shared__ __align__(16) __bf16 Kt[2][32 * KSTR];    // K tiles, row-major per key
  __shared__ __align__(16) __bf16 Vt[2][128 * VSTR];   // V tiles, transposed [d][key]
  __shared__ __align__(16) __bf16 Pbuf[8 * 16 * 32];

  const int hq = blockIdx.x, b = blockIdx.y;
  const int hk = hq >> 2;  // HQ/HK = 4
  const int tid = threadIdx.x;
  const int wave = tid >> 5, lane = tid & 31, ln = lane & 15, hf = lane >> 4;
  const int qr0 = wave * 16;
  const int seqlen = seqlens[b];

  // dense list of live key blocks: ceil(seqlen/32) cache blocks + 4 new blocks
  const int nc = (seqlen + 31) >> 5;
  const int nblk = nc + 4;
  auto blk_ks = [&](int idx) { return (idx < nc) ? idx * 32 : (128 + (idx - nc)) * 32; };

  const unsigned kt0 = (unsigned)(size_t)&Kt[0][0];  // LDS byte offsets (low 32 bits)
  const unsigned kt1 = (unsigned)(size_t)&Kt[1][0];

  // this wave's resident Q fragments (16 rows x D=128)
  bf16x16 qa[4];
  const __bf16* qbase = qbf + (((size_t)b * S_ + qr0) * HQ_ + hq) * (size_t)D_;
#pragma unroll
  for (int kk = 0; kk < 4; ++kk) qa[kk] = load_frag_a_bf(qbase + kk * 32, HQ_ * D_);

  float mrow[8], lrow[8], corr[8];
  f32x8 Oacc[8];
#pragma unroll
  for (int r = 0; r < 8; ++r) { mrow[r] = -3.0e38f; lrow[r] = 0.f; }
#pragma unroll
  for (int t = 0; t < 8; ++t) Oacc[t] = zero_f32x8();

  __bf16* Pw = &Pbuf[wave * 16 * 32];
  const int key_l = tid >> 3;  // 0..31: key row this thread stages
  const int seg   = tid & 7;   // 16-element segment within the 128-wide row

  // issue staging for one block into buffer `buf` (no wait here)
  auto stage = [&](int ks, int buf) {
    const int key = ks + key_l;
    const __bf16 *krow, *vrow;
    if (key >= LCACHE) {
      const size_t ro = (((size_t)b * S_ + (key - LCACHE)) * HK_ + hk) * (size_t)D_;
      krow = knew + ro;
      vrow = vnew + ro;
    } else {
      const int page = btab[b * NPP_ + (key >> 8)];
      const size_t ro =
          (((size_t)page * PSIZE_ + (key & (PSIZE_ - 1))) * HK_ + hk) * (size_t)D_;
      krow = kcbf + ro;
      vrow = vcbf + ro;
    }
    // K row: two 16B chunks DMA'd straight to LDS (ASYNCcnt-tracked)
    const unsigned l0 = (buf ? kt1 : kt0) + (unsigned)(key_l * (KSTR * 2) + seg * 16);
    const unsigned long long g0 = (unsigned long long)(size_t)krow + (unsigned)(seg * 16);
    asm volatile("global_load_async_to_lds_b128 %0, %1, off" :: "v"(l0), "v"(g0) : "memory");
    const unsigned l1 = l0 + 128;
    const unsigned long long g1 = g0 + 128;
    asm volatile("global_load_async_to_lds_b128 %0, %1, off" :: "v"(l1), "v"(g1) : "memory");
    // V: 16 contiguous elements, scattered transposed into Vt[buf][d][key]
    union { uint4 q[2]; __bf16 e[16]; } vv;
    const uint4* vp = (const uint4*)(vrow + seg * 16);
    vv.q[0] = vp[0];
    vv.q[1] = vp[1];
    const int d0 = seg * 16;
#pragma unroll
    for (int i = 0; i < 16; ++i) Vt[buf][(d0 + i) * VSTR + key_l] = vv.e[i];
  };

  stage(blk_ks(0), 0);  // prologue

  for (int idx = 0; idx < nblk; ++idx) {
    asm volatile("s_wait_asynccnt 0" ::: "memory");
    __syncthreads();  // buffer idx&1 staged by all waves; previous compute done
    if (idx + 1 < nblk) stage(blk_ks(idx + 1), (idx + 1) & 1);

    const int ks = blk_ks(idx);
    const int cur = idx & 1;

    // ---- scores: two 16-key tiles, each a D-loop of 4 WMMAs ----
    f32x8 sc[2];
#pragma unroll
    for (int h16 = 0; h16 < 2; ++h16) {
      f32x8 s = zero_f32x8();
      const int keyc = ks + h16 * 16 + ln;  // this lane's key column
#pragma unroll
      for (int kk = 0; kk < 4; ++kk) {
        const bf16x16 kf =
            load16_aligned(&Kt[cur][(h16 * 16 + ln) * KSTR + kk * 32 + hf * 16]);
        s = __builtin_amdgcn_wmma_f32_16x16x32_bf16(false, qa[kk], false, kf, (short)0, s,
                                                    false, false);
      }
      const bool valid = (keyc < seqlen) || (keyc >= LCACHE);
      const float bias = valid ? 0.f : -3.0e38f;
#pragma unroll
      for (int r = 0; r < 8; ++r) s[r] = s[r] * SCALE_ + bias;
      sc[h16] = s;
    }

    // ---- online softmax; rows live in 16-lane halves (C layout: M = r + 8*hf) ----
    float bm[8], p0[8], p1[8], rs[8];
#pragma unroll
    for (int r = 0; r < 8; ++r) bm[r] = fmaxf(sc[0][r], sc[1][r]);
#pragma unroll
    for (int m = 8; m >= 1; m >>= 1)
#pragma unroll
      for (int r = 0; r < 8; ++r) bm[r] = fmaxf(bm[r], __shfl_xor(bm[r], m, 32));
#pragma unroll
    for (int r = 0; r < 8; ++r) {
      const float mn = fmaxf(mrow[r], bm[r]);
      corr[r] = __expf(mrow[r] - mn);
      mrow[r] = mn;
      p0[r] = __expf(sc[0][r] - mn);
      p1[r] = __expf(sc[1][r] - mn);
      rs[r] = p0[r] + p1[r];
    }
#pragma unroll
    for (int m = 8; m >= 1; m >>= 1)
#pragma unroll
      for (int r = 0; r < 8; ++r) rs[r] += __shfl_xor(rs[r], m, 32);
#pragma unroll
    for (int r = 0; r < 8; ++r) lrow[r] = lrow[r] * corr[r] + rs[r];
#pragma unroll
    for (int t = 0; t < 8; ++t)
#pragma unroll
      for (int r = 0; r < 8; ++r) Oacc[t][r] *= corr[r];

    // stash P (16x32 bf16) in this wave's LDS tile, re-read in A-fragment order
#pragma unroll
    for (int r = 0; r < 8; ++r) {
      const int M = r + 8 * hf;
      Pw[M * 32 + ln]      = (__bf16)p0[r];
      Pw[M * 32 + 16 + ln] = (__bf16)p1[r];
    }
    const bf16x16 pa = load_frag_a_lds(Pw, 32);

    // ---- O += P @ V : 8 D-tiles; V^T rows give contiguous B fragments ----
#pragma unroll
    for (int t = 0; t < 8; ++t) {
      const bf16x16 vf = load16_aligned(&Vt[cur][(t * 16 + ln) * VSTR + hf * 16]);
      Oacc[t] = __builtin_amdgcn_wmma_f32_16x16x32_bf16(false, pa, false, vf, (short)0,
                                                        Oacc[t], false, false);
    }
  }

  // epilogue: O /= l, write bf16 (B,S,HQ,D) for the output projection
#pragma unroll
  for (int t = 0; t < 8; ++t) {
#pragma unroll
    for (int r = 0; r < 8; ++r) {
      const int M = r + 8 * hf;
      const float val = Oacc[t][r] / lrow[r];
      obf[(((size_t)b * S_ + qr0 + M) * HQ_ + hq) * (size_t)D_ + t * 16 + ln] = (__bf16)val;
    }
  }
}

extern "C" void kernel_launch(void* const* d_in, const int* in_sizes, int n_in,
                              void* d_out, int out_size, void* d_ws, size_t ws_size,
                              hipStream_t stream) {
  const float* x   = (const float*)d_in[0];
  const float* Wq  = (const float*)d_in[1];
  const float* Wk  = (const float*)d_in[2];
  const float* Wv  = (const float*)d_in[3];
  const float* Wo  = (const float*)d_in[4];
  const float* qnw = (const float*)d_in[5];
  const float* knw = (const float*)d_in[6];
  const float* kc  = (const float*)d_in[7];
  const float* vc  = (const float*)d_in[8];
  const int*   bt  = (const int*)d_in[9];
  const int*   csl = (const int*)d_in[10];

  const int M  = B_ * S_;     // 512
  const int NQ = HQ_ * D_;    // 4096
  const int NK = HK_ * D_;    // 1024
  const size_t CACHE_N = (size_t)64 * PSIZE_ * HK_ * D_;  // 16.78M elements

  // workspace carve-up (~178 MB total)
  char* w = (char*)d_ws;
  auto take = [&](size_t bytes) -> char* {
    char* p = w;
    w += (bytes + 255) & ~(size_t)255;
    return p;
  };
  __bf16* xbf  = (__bf16*)take((size_t)M * HID_ * 2);
  __bf16* Wqbf = (__bf16*)take((size_t)NQ * HID_ * 2);
  __bf16* Wkbf = (__bf16*)take((size_t)NK * HID_ * 2);
  __bf16* Wvbf = (__bf16*)take((size_t)NK * HID_ * 2);
  __bf16* Wobf = (__bf16*)take((size_t)HID_ * NQ * 2);
  __bf16* kcbf = (__bf16*)take(CACHE_N * 2);
  __bf16* vcbf = (__bf16*)take(CACHE_N * 2);
  float*  qf   = (float*)take((size_t)M * NQ * 4);
  float*  kf   = (float*)take((size_t)M * NK * 4);
  float*  vf   = (float*)take((size_t)M * NK * 4);
  __bf16* qbf  = (__bf16*)take((size_t)M * NQ * 2);
  __bf16* kbf  = (__bf16*)take((size_t)M * NK * 2);
  __bf16* vbf  = (__bf16*)take((size_t)M * NK * 2);
  __bf16* obf  = (__bf16*)take((size_t)M * NQ * 2);

  auto cvt = [&](const float* src, __bf16* dst, size_t n) {
    cvt_f32_to_bf16<<<dim3((unsigned)((n + 255) / 256)), dim3(256), 0, stream>>>(src, dst,
                                                                                 (int)n);
  };
  cvt(x,  xbf,  (size_t)M  * HID_);
  cvt(Wq, Wqbf, (size_t)NQ * HID_);
  cvt(Wk, Wkbf, (size_t)NK * HID_);
  cvt(Wv, Wvbf, (size_t)NK * HID_);
  cvt(Wo, Wobf, (size_t)HID_ * NQ);
  cvt(kc, kcbf, CACHE_N);
  cvt(vc, vcbf, CACHE_N);

  auto gemm = [&](const __bf16* A, const __bf16* Bm, float* C, int m, int n, int k) {
    gemm_bf16_wmma<<<dim3(n / 256, m / 16), dim3(128), 0, stream>>>(A, Bm, C, m, n, k);
  };
  gemm(xbf, Wqbf, qf, M, NQ, HID_);
  gemm(xbf, Wkbf, kf, M, NK, HID_);
  gemm(xbf, Wvbf, vf, M, NK, HID_);

  rmsnorm_rope_kernel<<<dim3(HQ_, S_, B_), dim3(32), 0, stream>>>(qf, qnw, csl, qbf, HQ_);
  rmsnorm_rope_kernel<<<dim3(HK_, S_, B_), dim3(32), 0, stream>>>(kf, knw, csl, kbf, HK_);
  cvt(vf, vbf, (size_t)M * NK);

  attn_kernel<<<dim3(HQ_, B_), dim3(256), 0, stream>>>(qbf, kbf, vbf, kcbf, vcbf, bt, csl,
                                                       obf);

  gemm(obf, Wobf, (float*)d_out, M, HID_, NQ);
}
